// GRU_26542897889421
// MI455X (gfx1250) — matrix operand order repaired and checked by reading, not compiled, not run on hardware
//
#include <hip/hip_runtime.h>
#include <cmath>

typedef __bf16 v16bf __attribute__((ext_vector_type(16)));
typedef float  v8f   __attribute__((ext_vector_type(8)));

constexpr int kT = 2048;
constexpr int kD = 256;
constexpr int kH = 256;
constexpr int SSTR = 264;   // bf16 row stride: 528 B = 132 dwords -> 4-bank skew/row

__device__ __forceinline__ unsigned short f2bf(float f) {
  unsigned u = __float_as_uint(f);
  u += 0x7FFFu + ((u >> 16) & 1u);          // round-to-nearest-even
  return (unsigned short)(u >> 16);
}

__device__ __forceinline__ float sigmoidf_(float v) {
  return 1.0f / (1.0f + __expf(-v));        // v_exp_f32 path
}

// A fragment (16x32 bf16, M=16): lane row m=llo; vgpr q holds K pair
// K = 2*(q&3) + 8*lhi + 16*(q>>2)  (pairs contiguous in row-major LDS).
__device__ __forceinline__ v16bf ldfragA(const unsigned short* rowp, int k0, int lhi) {
  union { unsigned u[8]; v16bf v; } F;
#pragma unroll
  for (int q = 0; q < 8; ++q) {
    const int ka = 2 * (q & 3) + 8 * lhi + 16 * (q >> 2);
    F.u[q] = *(const unsigned*)(rowp + k0 + ka);
  }
  return F.v;
}

// B fragment (32x16 bf16): lane col n=llo; vgpr q holds K pair
// K = 2*(q&3) + 8*lhi + 16*(q>>2). Row-major weight slice [n][k] is exactly W^T as WMMA wants.
__device__ __forceinline__ v16bf ldfragB(const unsigned short* rowp, int k0, int lhi) {
  union { unsigned u[8]; v16bf v; } F;
#pragma unroll
  for (int q = 0; q < 8; ++q)
    F.u[q] = *(const unsigned*)(rowp + k0 + 2 * (q & 3) + 8 * lhi + 16 * (q >> 2));
  return F.v;
}

// One 16x16 tile over a K=128 half: 4 WMMAs in two independent chains.
__device__ __forceinline__ void tile_k128(const unsigned short* Arow,
                                          const unsigned short* Brow,
                                          int kbase, int lhi, int llo,
                                          float* dst /* [16][17] */) {
  v8f a0 = {0.f, 0.f, 0.f, 0.f, 0.f, 0.f, 0.f, 0.f};
  v8f a1 = {0.f, 0.f, 0.f, 0.f, 0.f, 0.f, 0.f, 0.f};
  a0 = __builtin_amdgcn_wmma_f32_16x16x32_bf16(false, ldfragA(Arow, kbase +  0, lhi),
                                               false, ldfragB(Brow, kbase +  0, lhi),
                                               (short)0, a0, false, false);
  a1 = __builtin_amdgcn_wmma_f32_16x16x32_bf16(false, ldfragA(Arow, kbase + 32, lhi),
                                               false, ldfragB(Brow, kbase + 32, lhi),
                                               (short)0, a1, false, false);
  a0 = __builtin_amdgcn_wmma_f32_16x16x32_bf16(false, ldfragA(Arow, kbase + 64, lhi),
                                               false, ldfragB(Brow, kbase + 64, lhi),
                                               (short)0, a0, false, false);
  a1 = __builtin_amdgcn_wmma_f32_16x16x32_bf16(false, ldfragA(Arow, kbase + 96, lhi),
                                               false, ldfragB(Brow, kbase + 96, lhi),
                                               (short)0, a1, false, false);
  // C/D layout: vgpr q -> row M = q + 8*lhi, col N = llo
#pragma unroll
  for (int q = 0; q < 8; ++q) dst[(q + 8 * lhi) * 17 + llo] = a0[q] + a1[q];
}

// Grid: 64 blocks x 192 threads (6 waves).
//   group = blockIdx.x>>4  -> batch rows [16g, 16g+16)
//   wg    = blockIdx.x&15  -> hidden cols [16*wg, 16*wg+16) (all 3 gates)
// Wave w -> (gate = w%3, K-half = w/3); each phase (x-proj / h-proj) runs all
// 6 waves; the two K-halves of each gate tile are summed in the combine step.
__global__ __launch_bounds__(192) void gru_scan(
    const float* __restrict__ x,          // [B,T,D]
    const float* __restrict__ h0,         // [1,B,H]
    const float* __restrict__ Wx,         // [3H,D]
    const float* __restrict__ bx,         // [3H]
    const float* __restrict__ Wh,         // [3H,H]
    const float* __restrict__ bh,         // [3H]
    float* __restrict__ out,              // [B,T,H]
    unsigned short* __restrict__ hbuf,    // [4 groups][2][16][256] bf16
    unsigned* __restrict__ cnt)           // [4 groups][T+1] arrival counters
{
  __shared__ unsigned short sWx[48 * SSTR];   // bf16 Wx slice [gate*16+j][k]
  __shared__ unsigned short sWh[48 * SSTR];   // bf16 Wh slice
  __shared__ unsigned short sH [16 * SSTR];   // bf16 h_{t-1} [m][k]
  __shared__ unsigned short sX [16 * SSTR];   // bf16 x_t     [m][k]
  __shared__ float sPX[6][16][17];            // x-proj tiles (gate, K-half)
  __shared__ float sPH[6][16][17];            // h-proj tiles
  __shared__ float sHprev[16 * 17];           // own fp32 h slice [m][j]
  __shared__ float sBx[48], sBh[48];

  const int tid   = threadIdx.x;
  const int group = blockIdx.x >> 4;
  const int b0    = group * 16;
  const int n0    = (blockIdx.x & 15) * 16;

  // ---- one-time: stage weight slices (f32 -> bf16) and biases into LDS ----
  for (int c = tid; c < 3072; c += 192) {     // 48 rows * 64 float4 chunks
    int r    = c >> 6;                        // 0..47 = gate*16 + j
    int col  = (c & 63) * 4;
    int gate = r >> 4, j = r & 15;
    long grow = (long)(gate * 256 + n0 + j) * kD + col;
    float4 fx = *(const float4*)(Wx + grow);
    float4 fh = *(const float4*)(Wh + grow);
    unsigned short* dx = sWx + r * SSTR + col;
    unsigned short* dh = sWh + r * SSTR + col;
    dx[0] = f2bf(fx.x); dx[1] = f2bf(fx.y); dx[2] = f2bf(fx.z); dx[3] = f2bf(fx.w);
    dh[0] = f2bf(fh.x); dh[1] = f2bf(fh.y); dh[2] = f2bf(fh.z); dh[3] = f2bf(fh.w);
  }
  for (int i = tid; i < 48; i += 192) {
    int gate = i >> 4, j = i & 15;
    sBx[i] = bx[gate * 256 + n0 + j];
    sBh[i] = bh[gate * 256 + n0 + j];
  }

  // ---- initial hidden: own fp32 slice to LDS, bf16 slice to ping buffer 0 ----
  {
    unsigned short* hb = hbuf + (group * 2 + 0) * 4096;
    for (int e = tid; e < 256; e += 192) {
      int m = e >> 4, j = e & 15;
      float hv = h0[(b0 + m) * kH + n0 + j];
      sHprev[m * 17 + j] = hv;
      hb[m * 256 + n0 + j] = f2bf(hv);
    }
  }
  __threadfence();
  __syncthreads();
  if (tid == 0)
    __hip_atomic_fetch_add(&cnt[group * (kT + 1)], 1u,
                           __ATOMIC_RELEASE, __HIP_MEMORY_SCOPE_AGENT);

  const int w     = tid >> 5;        // wave 0..5
  const int gate  = w % 3;           // 0=r,1=u,2=n
  const int kbase = (w / 3) * 128;   // K-half
  const int lane  = tid & 31;
  const int llo   = lane & 15;
  const int lhi   = lane >> 4;

  const unsigned short* ArowX = sX + llo * SSTR;
  const unsigned short* ArowH = sH + llo * SSTR;
  const unsigned short* BrowX = sWx + (gate * 16 + llo) * SSTR;
  const unsigned short* BrowH = sWh + (gate * 16 + llo) * SSTR;

  for (int t = 0; t < kT; ++t) {
    // ================= X phase (independent of h_{t-1}) =================
    for (int c = tid; c < 1024; c += 192) {   // 16 rows * 64 float4 chunks
      int row = c >> 6, col = (c & 63) * 4;
      const float* px = x + (long)(b0 + row) * (kT * kD) + (long)t * kD + col;
      float4 g = *(const float4*)px;
      unsigned short* dxx = sX + row * SSTR + col;
      dxx[0] = f2bf(g.x); dxx[1] = f2bf(g.y); dxx[2] = f2bf(g.z); dxx[3] = f2bf(g.w);
      if (t + 1 < kT) __builtin_prefetch(px + kD, 0, 3);   // next x row, WGP scope
    }
    __syncthreads();
    tile_k128(ArowX, BrowX, kbase, lhi, llo, &sPX[w][0][0]);

    // ================= wait for h_{t-1} from peer WGs =================
    const unsigned cidx = (unsigned)(group * (kT + 1) + t);
    if (tid == 0) {
      // relaxed spin (no per-poll cache invalidate) ...
      while (__hip_atomic_load(&cnt[cidx], __ATOMIC_RELAXED,
                               __HIP_MEMORY_SCOPE_AGENT) < 16u)
        __builtin_amdgcn_s_sleep(1);
      // ... then a single acquire to pair with the peers' release
      (void)__hip_atomic_load(&cnt[cidx], __ATOMIC_ACQUIRE,
                              __HIP_MEMORY_SCOPE_AGENT);
    }
    __syncthreads();

    // ---- async-copy bf16 h tile (8 KB) global -> LDS with row skew.
    //      DEV scope: force-miss WGP caches, read peers' data from L2. ----
    {
      const unsigned short* hsrcb = hbuf + (group * 2 + (t & 1)) * 4096;
      for (int c = tid; c < 512; c += 192) {  // 16 rows * 32 x 16B chunks
        int row = c >> 5, col8 = (c & 31) * 8;
        unsigned lds = (unsigned)(size_t)(sH + row * SSTR + col8);
        unsigned long long ga =
            (unsigned long long)(size_t)(hsrcb + row * 256 + col8);
        asm volatile("global_load_async_to_lds_b128 %0, %1, off scope:SCOPE_DEV"
                     :: "v"(lds), "v"(ga) : "memory");
      }
      asm volatile("s_wait_asynccnt 0x0" ::: "memory");
    }
    __syncthreads();

    // ================= H phase =================
    tile_k128(ArowH, BrowH, kbase, lhi, llo, &sPH[w][0][0]);
    __syncthreads();

    // ================= gate math (fp32), publish =================
    unsigned short* hdstb = hbuf + (group * 2 + ((t + 1) & 1)) * 4096;
    for (int e = tid; e < 256; e += 192) {
      int m = e >> 4, j = e & 15;
      float xr = sPX[0][m][j] + sPX[3][m][j];
      float xu = sPX[1][m][j] + sPX[4][m][j];
      float xn = sPX[2][m][j] + sPX[5][m][j];
      float hr = sPH[0][m][j] + sPH[3][m][j];
      float hu = sPH[1][m][j] + sPH[4][m][j];
      float hn = sPH[2][m][j] + sPH[5][m][j];
      float r  = sigmoidf_(xr + hr + sBx[j]      + sBh[j]);
      float u  = sigmoidf_(xu + hu + sBx[16 + j] + sBh[16 + j]);
      float nn = tanhf(xn + sBx[32 + j] + r * (hn + sBh[32 + j]));
      float hp = sHprev[m * 17 + j];          // own fp32 slice (exact)
      float hy = u * hp + (1.0f - u) * nn;
      out[(long)(b0 + m) * (kT * kH) + (long)t * kH + n0 + j] = hy;
      sHprev[m * 17 + j] = hy;                // same thread owns this element
      hdstb[m * 256 + n0 + j] = f2bf(hy);
    }
    __threadfence();
    __syncthreads();
    if (tid == 0)
      __hip_atomic_fetch_add(&cnt[cidx + 1], 1u,
                             __ATOMIC_RELEASE, __HIP_MEMORY_SCOPE_AGENT);
  }
}

extern "C" void kernel_launch(void* const* d_in, const int* in_sizes, int n_in,
                              void* d_out, int out_size, void* d_ws, size_t ws_size,
                              hipStream_t stream) {
  (void)in_sizes; (void)n_in; (void)out_size; (void)ws_size;
  const float* x  = (const float*)d_in[0];
  const float* h0 = (const float*)d_in[1];
  const float* Wx = (const float*)d_in[2];
  const float* bx = (const float*)d_in[3];
  const float* Wh = (const float*)d_in[4];
  const float* bh = (const float*)d_in[5];
  float* out = (float*)d_out;

  // ws layout: [4][2][16][256] bf16 h ping-pong (64 KB) + [4][T+1] u32 counters
  unsigned short* hbuf = (unsigned short*)d_ws;
  unsigned* cnt = (unsigned*)((char*)d_ws + 4 * 2 * 16 * 256 * sizeof(unsigned short));
  const size_t zero_bytes = 4 * 2 * 16 * 256 * sizeof(unsigned short)
                          + 4 * (kT + 1) * sizeof(unsigned);
  hipMemsetAsync(d_ws, 0, zero_bytes, stream);   // reset counters every call

  hipLaunchKernelGGL(gru_scan, dim3(64), dim3(192), 0, stream,
                     x, h0, Wx, bx, Wh, bh, out, hbuf, cnt);
}
